// AstraloraLayer_17162689314818
// MI455X (gfx1250) — matrix operand order repaired and checked by reading, not compiled, not run on hardware
//
#include <hip/hip_runtime.h>

// y = scale * (x @ W^T) + x   |  x:[16384,1024] f32, W:[1024,1024] f32 (row-major, K contiguous)
// bf16 WMMA GEMM, f32 accumulate, residual in f32.
// Workgroup tile 128x256xK32, 8 wave32s as 2(M) x 4(N), 64x64 per wave -> 16 WMMA / 8 frag-loads per K-step.
// Double-buffered LDS, on-the-fly f32->bf16 via v_cvt_pk_bf16_f32, global_prefetch for K+2.

typedef __attribute__((ext_vector_type(16))) __bf16 v16bf;
typedef __attribute__((ext_vector_type(8)))  __bf16 v8bf;
typedef __attribute__((ext_vector_type(8)))  float  v8f;
typedef __attribute__((ext_vector_type(4)))  float  v4f;

#define D_INP    1024
#define D_OUT    1024
#define N_TOKENS 16384

#define BM 128
#define BN 256
#define BK 32
#define LDS_STRIDE 40              // 32 K-elems + 8 pad (bf16): 80B rows, 16B-aligned, bank-spread
#define NSTEPS (D_INP / BK)        // 32 K-steps

__global__ __launch_bounds__(256)
void astralora_matvec_scale_res(const float* __restrict__ x,
                                const float* __restrict__ w,
                                const float* __restrict__ scale_p,
                                float* __restrict__ out)
{
    __shared__ __bf16 ldsA[2][BM * LDS_STRIDE];   // 2 x 10 KB
    __shared__ __bf16 ldsB[2][BN * LDS_STRIDE];   // 2 x 20 KB

    const int tid  = threadIdx.x;
    const int lane = tid & 31;
    const int wave = tid >> 5;
    const int l16  = lane & 15;
    const int hh   = lane >> 4;          // K-striping half-select
    const int wm   = (wave & 1) * 64;    // 2 waves tile M (4 x 16)
    const int wn   = (wave >> 1) * 64;   // 4 waves tile N (4 x 16)

    const int m0 = blockIdx.y * BM;
    const int n0 = blockIdx.x * BN;

    // staging: A -> thread owns a 16-f32 half-row; B -> thread owns a full 32-f32 row
    const int srowA = tid >> 1;          // 0..127
    const int scolA = (tid & 1) * 16;    // 0 or 16
    const float* gA = x + (size_t)(m0 + srowA) * D_INP + scolA;
    const float* gB = w + (size_t)(n0 + tid) * D_INP;

    v8f acc[4][4];
    #pragma unroll
    for (int i = 0; i < 4; ++i)
        #pragma unroll
        for (int j = 0; j < 4; ++j)
            acc[i][j] = v8f{};

    v4f ra[4];                           // 16 f32 of A panel
    v4f rb[8];                           // 32 f32 of B panel

    auto stage_load = [&](int k) {
        const float* pa = gA + k * BK;
        const float* pb = gB + k * BK;
        #pragma unroll
        for (int i = 0; i < 4; ++i) ra[i] = *(const v4f*)(pa + 4 * i);
        #pragma unroll
        for (int i = 0; i < 8; ++i) rb[i] = *(const v4f*)(pb + 4 * i);
    };

    auto stage_store = [&](int buf) {
        __bf16* da = &ldsA[buf][srowA * LDS_STRIDE + scolA];
        __bf16* db = &ldsB[buf][tid * LDS_STRIDE];
        v8bf pa0, pa1;
        #pragma unroll
        for (int i = 0; i < 4; ++i) {
            pa0[i]     = (__bf16)ra[0][i];
            pa0[i + 4] = (__bf16)ra[1][i];
            pa1[i]     = (__bf16)ra[2][i];
            pa1[i + 4] = (__bf16)ra[3][i];
        }
        *(v8bf*)(da)     = pa0;          // 16B ds_store
        *(v8bf*)(da + 8) = pa1;
        #pragma unroll
        for (int c = 0; c < 4; ++c) {
            v8bf pb;
            #pragma unroll
            for (int i = 0; i < 4; ++i) {
                pb[i]     = (__bf16)rb[2 * c][i];
                pb[i + 4] = (__bf16)rb[2 * c + 1][i];
            }
            *(v8bf*)(db + 8 * c) = pb;   // 4 x 16B ds_store
        }
    };

    auto compute = [&](int buf) {
        // A frag (16x32 bf16): lane L (M=L&15, h=L>>4) holds K=[16h..16h+15] contiguous
        v16bf afrag[4];
        #pragma unroll
        for (int mt = 0; mt < 4; ++mt) {
            const __bf16* p = &ldsA[buf][(wm + mt * 16 + l16) * LDS_STRIDE + hh * 16];
            v8bf lo = *(const v8bf*)(p);        // K = 16h .. 16h+7
            v8bf hi = *(const v8bf*)(p + 8);    // K = 16h+8 .. 16h+15
            afrag[mt] = __builtin_shufflevector(lo, hi,
                0, 1, 2, 3, 4, 5, 6, 7, 8, 9, 10, 11, 12, 13, 14, 15);
        }
        // B frag (32x16 bf16): lane L (N=L&15, h=L>>4) holds K=[16h..16h+15] contiguous
        v16bf bfrag[4];
        #pragma unroll
        for (int nt = 0; nt < 4; ++nt)
            bfrag[nt] = *(const v16bf*)(&ldsB[buf][(wn + nt * 16 + l16) * LDS_STRIDE + hh * 16]);

        #pragma unroll
        for (int mt = 0; mt < 4; ++mt)
            #pragma unroll
            for (int nt = 0; nt < 4; ++nt)
                acc[mt][nt] = __builtin_amdgcn_wmma_f32_16x16x32_bf16(
                    /*neg_a=*/false, afrag[mt],
                    /*neg_b=*/false, bfrag[nt],
                    /*c_mod=*/(short)0, acc[mt][nt],
                    /*reuse_a=*/false, /*reuse_b=*/false);
    };

    // ---- prologue: fill buffer 0
    stage_load(0);
    stage_store(0);
    __syncthreads();

    // ---- main pipeline: global load (k+1) -> compute (k) -> cvt+LDS store (k+1) -> barrier
    for (int k = 0; k < NSTEPS; ++k) {
        const int cur = k & 1;
        if (k + 1 < NSTEPS) {
            stage_load(k + 1);
            if (k + 2 < NSTEPS) {   // pull K+2 panel toward the WGP caches
                __builtin_prefetch(gA + (k + 2) * BK, 0, 3);
                __builtin_prefetch(gB + (k + 2) * BK, 0, 3);
            }
        }
        compute(cur);
        if (k + 1 < NSTEPS)
            stage_store(cur ^ 1);
        __syncthreads();
    }

    // ---- epilogue: y = acc*scale + x  (C/D layout: VGPR r -> M=r (lanes 0-15), M=r+8 (lanes 16-31))
    const float s = scale_p[0];
    #pragma unroll
    for (int mt = 0; mt < 4; ++mt) {
        #pragma unroll
        for (int nt = 0; nt < 4; ++nt) {
            const int col = n0 + wn + nt * 16 + l16;
            #pragma unroll
            for (int r = 0; r < 8; ++r) {
                const int row = m0 + wm + mt * 16 + r + 8 * hh;
                const size_t idx = (size_t)row * D_OUT + col;
                out[idx] = acc[mt][nt][r] * s + x[idx];
            }
        }
    }
}

extern "C" void kernel_launch(void* const* d_in, const int* in_sizes, int n_in,
                              void* d_out, int out_size, void* d_ws, size_t ws_size,
                              hipStream_t stream) {
    const float* x  = (const float*)d_in[0];
    const float* w  = (const float*)d_in[1];
    const float* sc = (const float*)d_in[2];
    float* out      = (float*)d_out;

    // grid: N-blocks fastest so the 4 WGs sharing an x M-panel co-run and hit L2 (W f32 is L2-resident)
    dim3 grid(D_OUT / BN, N_TOKENS / BM);   // (4, 128)
    astralora_matvec_scale_res<<<grid, 256, 0, stream>>>(x, w, sc, out);
}
